// _MultiheadAttention_fu_455266533976
// MI455X (gfx1250) — compile-verified, matrix operand order checked by reading, not run on hardware
//
#include <hip/hip_runtime.h>
#include <hip/hip_bf16.h>

typedef __attribute__((ext_vector_type(16))) __bf16 bf16x16;
typedef __attribute__((ext_vector_type(8)))  __bf16 bf16x8;
typedef __attribute__((ext_vector_type(8)))  float  f32x8;

constexpr float SCALE_ = 0.04419417382415922f;  // 512^-0.5
constexpr int   LDSTR  = 72;                    // 144B row stride: 16B-aligned ds ops, bank-skewed

// ---------------------------------------------------------------------------
// Packing kernels (fp32 -> bf16, complex-fold into GEMM operands)
// ---------------------------------------------------------------------------

// X = [Qr | Qi]  -> [32768, 1024] bf16
__global__ __launch_bounds__(256) void pack_x(const float* __restrict__ Qr,
                                              const float* __restrict__ Qi,
                                              __bf16* __restrict__ X) {
  size_t idx = (size_t)blockIdx.x * 256 + threadIdx.x;   // 32768*1024
  int k = idx & 1023;
  size_t m = idx >> 10;
  float v = (k < 512) ? Qr[m * 512 + k] : Qi[m * 512 + (k - 512)];
  X[idx] = (__bf16)v;
}

// Wproj Bt-layout [3072 n][1024 k]; 6 groups of 512 cols: qr,qi,kr1,ki1,vr1,vi1
__global__ __launch_bounds__(256) void pack_wproj(
    const float* WQ, const float* bQ, const float* WQi, const float* bQi,
    const float* WK, const float* bK, const float* WKi, const float* bKi,
    const float* WV, const float* bV, const float* WVi, const float* bVi,
    __bf16* __restrict__ Bt, float* __restrict__ bias) {
  size_t idx = (size_t)blockIdx.x * 256 + threadIdx.x;   // 3072*1024
  int k = idx & 1023;
  int n = (int)(idx >> 10);
  int g = n >> 9, j = n & 511, kk = k & 511;
  bool hi = (k >= 512);
  const float *W1, *W2; float s2;
  switch (g) {
    case 0:  W1 = WQ;  W2 = WQi; s2 = -1.f; break;   // qr
    case 1:  W1 = WQi; W2 = WQ;  s2 =  1.f; break;   // qi
    case 2:  W1 = WK;  W2 = WKi; s2 = -1.f; break;   // kr1
    case 3:  W1 = WKi; W2 = WK;  s2 =  1.f; break;   // ki1
    case 4:  W1 = WV;  W2 = WVi; s2 = -1.f; break;   // vr1
    default: W1 = WVi; W2 = WV;  s2 =  1.f; break;   // vi1
  }
  float v = hi ? s2 * W2[j * 512 + kk] : W1[j * 512 + kk];
  Bt[idx] = (__bf16)v;
  if (idx < 3072) {
    int nn = (int)idx, gg = nn >> 9, jj = nn & 511;
    const float* bA = (gg < 2) ? bQ  : (gg < 4) ? bK  : bV;
    const float* bB = (gg < 2) ? bQi : (gg < 4) ? bKi : bVi;
    bias[nn] = (gg & 1) ? (bA[jj] + bB[jj]) : (bA[jj] - bB[jj]);
  }
}

// Bt4 [1024 n][1024 k]: n<512 -> final_r col j (W | -W); n>=512 -> final_i (W | W)
__global__ __launch_bounds__(256) void pack_wout(const float* __restrict__ W,
                                                 const float* __restrict__ b,
                                                 __bf16* __restrict__ Bt,
                                                 float* __restrict__ bias) {
  size_t idx = (size_t)blockIdx.x * 256 + threadIdx.x;   // 1024*1024
  int k = idx & 1023;
  int n = (int)(idx >> 10);
  int j = n & 511, kk = k & 511;
  float v = W[j * 512 + kk];
  if (n < 512 && k >= 512) v = -v;
  Bt[idx] = (__bf16)v;
  if (idx < 1024) bias[idx] = (idx < 512) ? 0.f : 2.f * b[idx - 512];
}

// ---------------------------------------------------------------------------
// Pool(4) + complex mix -> KK (Bt [1024 n][512 k] per batch) and
// VV (Bt [1024 n][1024 k] per batch) with out_r/out_i signs folded in.
// ---------------------------------------------------------------------------
__global__ __launch_bounds__(256) void pool_combine(const __bf16* __restrict__ PROJ,
                                                    __bf16* __restrict__ KK,
                                                    __bf16* __restrict__ VV) {
  size_t idx = (size_t)blockIdx.x * 256 + threadIdx.x;   // 16*512*512
  int d = idx & 511;
  int t = (int)((idx >> 9) & 511);
  int b = (int)(idx >> 18);
  const __bf16* base = PROJ + ((size_t)(b * 2048 + t * 4)) * 3072 + 1024;
  float pk = 0, pki = 0, pv = 0, pvi = 0;
#pragma unroll
  for (int s = 0; s < 4; ++s) {
    const __bf16* row = base + (size_t)s * 3072;
    pk  += (float)row[d];
    pki += (float)row[512 + d];
    pv  += (float)row[1024 + d];
    pvi += (float)row[1536 + d];
  }
  pk *= 0.25f; pki *= 0.25f; pv *= 0.25f; pvi *= 0.25f;
  float kr = pk - pki, ki = pki + pk;
  float vr = pv - pvi, vi = pvi + pv;
  size_t kb = (size_t)b * 1024 * 512;
  KK[kb + (size_t)t * 512 + d]         = (__bf16)kr;
  KK[kb + (size_t)(512 + t) * 512 + d] = (__bf16)ki;
  size_t vb = (size_t)b * 1024 * 1024;
  VV[vb + (size_t)d * 1024 + t]               = (__bf16)vr;    // out_r:  P·vr
  VV[vb + (size_t)d * 1024 + 512 + t]         = (__bf16)(-vi); // out_r: -S·vi
  VV[vb + (size_t)(512 + d) * 1024 + t]       = (__bf16)vi;    // out_i:  P·vi
  VV[vb + (size_t)(512 + d) * 1024 + 512 + t] = (__bf16)vr;    // out_i: +S·vr
}

// ---------------------------------------------------------------------------
// Generic bf16 WMMA GEMM: C[M,N] = A[M,K] * Bt[N,K]^T (+bias).
// Tile 128x128x64, 8 wave32s (4 along M x 2 along N), wave tile 32x64,
// 16 v_wmma_f32_16x16x32_bf16 per barrier pair.
// ---------------------------------------------------------------------------
struct GemmParams {
  const __bf16* A;
  const __bf16* B;
  float*  outF;
  float*  outF2;
  __bf16* outH;
  const float* bias;
  int lda, ldb, ldc, K, zdiv, mode;
  long long aZ1, aZ2, bZ1, bZ2, cZ1, cZ2;
};

__device__ __forceinline__ bf16x16 load_frag(const __bf16* lds, int row0, int kofs, int lane) {
  // 16-bit A/B fragment per ISA 7.12.2: lane&15 = row, lane>>4 selects K halves
  const __bf16* p = lds + (size_t)(row0 + (lane & 15)) * LDSTR + kofs + ((lane >> 4) << 3);
  bf16x8 lo = *(const bf16x8*)p;          // K = base+0..7
  bf16x8 hi = *(const bf16x8*)(p + 16);   // K = base+16..23
  bf16x16 r;
#pragma unroll
  for (int e = 0; e < 8; ++e) { r[e] = lo[e]; r[e + 8] = hi[e]; }
  return r;
}

__global__ __launch_bounds__(256) void gemm_bf16_wmma(GemmParams p) {
  __shared__ __bf16 As[128 * LDSTR];
  __shared__ __bf16 Bs[128 * LDSTR];

  const int t = threadIdx.x;
  const int lane = t & 31;
  const int wid = t >> 5;
  const int waveM = wid >> 1;            // 0..3
  const int waveN = wid & 1;             // 0..1
  const int z = blockIdx.z;
  const int rowBase = blockIdx.y * 128;
  const int colBase = blockIdx.x * 128;

  const long long zq = z / p.zdiv, zr = z % p.zdiv;
  const __bf16* Ag = p.A + zq * p.aZ1 + zr * p.aZ2;
  const __bf16* Bg = p.B + zq * p.bZ1 + zr * p.bZ2;

  f32x8 Cacc[2][4] = {};

  const int r = t >> 1;                  // 0..127 rows of the tile
  const int c = (t & 1) << 5;            // 0 or 32 (bf16 elems)
  const __bf16* aSrc = Ag + (size_t)(rowBase + r) * p.lda + c;
  const __bf16* bSrc = Bg + (size_t)(colBase + r) * p.ldb + c;
  __bf16* aDst = &As[r * LDSTR + c];
  __bf16* bDst = &Bs[r * LDSTR + c];

  for (int kt = 0; kt < p.K; kt += 64) {
    uint4 a0 = *(const uint4*)(aSrc + kt);
    uint4 a1 = *(const uint4*)(aSrc + kt + 8);
    uint4 a2 = *(const uint4*)(aSrc + kt + 16);
    uint4 a3 = *(const uint4*)(aSrc + kt + 24);
    uint4 b0 = *(const uint4*)(bSrc + kt);
    uint4 b1 = *(const uint4*)(bSrc + kt + 8);
    uint4 b2 = *(const uint4*)(bSrc + kt + 16);
    uint4 b3 = *(const uint4*)(bSrc + kt + 24);
    if (kt + 64 < p.K) {                 // global_prefetch next K tile
      __builtin_prefetch(aSrc + kt + 64, 0, 0);
      __builtin_prefetch(bSrc + kt + 64, 0, 0);
    }
    __syncthreads();
    *(uint4*)aDst        = a0;
    *(uint4*)(aDst + 8)  = a1;
    *(uint4*)(aDst + 16) = a2;
    *(uint4*)(aDst + 24) = a3;
    *(uint4*)bDst        = b0;
    *(uint4*)(bDst + 8)  = b1;
    *(uint4*)(bDst + 16) = b2;
    *(uint4*)(bDst + 24) = b3;
    __syncthreads();

#pragma unroll
    for (int kc = 0; kc < 64; kc += 32) {
      bf16x16 af[2], bfr[4];
#pragma unroll
      for (int i = 0; i < 2; ++i) af[i] = load_frag(As, waveM * 32 + i * 16, kc, lane);
#pragma unroll
      for (int j = 0; j < 4; ++j) bfr[j] = load_frag(Bs, waveN * 64 + j * 16, kc, lane);
#pragma unroll
      for (int i = 0; i < 2; ++i)
#pragma unroll
        for (int j = 0; j < 4; ++j)
          Cacc[i][j] = __builtin_amdgcn_wmma_f32_16x16x32_bf16(
              false, af[i], false, bfr[j], (short)0, Cacc[i][j], false, false);
    }
  }

  // Epilogue. C frag layout: VGPR rr, lane -> M = rr + 8*(lane>>4), N = lane&15
  const long long cz = zq * p.cZ1 + zr * p.cZ2;
#pragma unroll
  for (int i = 0; i < 2; ++i) {
    int m0 = rowBase + waveM * 32 + i * 16 + ((lane >> 4) << 3);
#pragma unroll
    for (int j = 0; j < 4; ++j) {
      int n = colBase + waveN * 64 + j * 16 + (lane & 15);
      f32x8 acc = Cacc[i][j];
#pragma unroll
      for (int rr = 0; rr < 8; ++rr) {
        int m = m0 + rr;
        float v = acc[rr];
        if (p.mode == 0) {                       // fp32 direct (logits)
          if (p.bias) v += p.bias[n];
          p.outF[cz + (long long)m * p.ldc + n] = v;
        } else if (p.mode == 1) {                // bf16 direct + bias (PROJ)
          p.outH[cz + (long long)m * p.ldc + n] = (__bf16)(v + p.bias[n]);
        } else if (p.mode == 2) {                // shuffle-scatter bf16 (out GEMM)
          int d = n & 511, which = n >> 9;       // m == q within batch
          int f = d * 2048 + m;
          int i2 = f >> 9, j2 = f & 511;
          p.outH[cz + (long long)i2 * 1024 + which * 512 + j2] = (__bf16)v;
        } else {                                 // fp32 split-col to d_out (final)
          v += p.bias[n];
          if (n < 512) p.outF[(long long)m * 512 + n] = v;
          else         p.outF2[(long long)m * 512 + (n - 512)] = v;
        }
      }
    }
  }
}

// ---------------------------------------------------------------------------
// Softmax over 4 logit segments + P/S combine; emits attn_r/attn_i (fp32 to
// d_out) and [P|S] bf16 for the output GEMM.
// ---------------------------------------------------------------------------
__global__ __launch_bounds__(256) void softmax_combine(const float* __restrict__ LOG,
                                                       float* __restrict__ attnR,
                                                       float* __restrict__ attnI,
                                                       __bf16* __restrict__ PS) {
  __shared__ float seg[4][512];
  __shared__ float red[256];
  const int t = threadIdx.x;
  const int b = blockIdx.x >> 11;
  const int q = blockIdx.x & 2047;
  const float* r0 = LOG + ((long long)(b * 2) * 2048 + q) * 1024;   // qr side: A|B
  const float* r1 = r0 + (long long)2048 * 1024;                    // qi side: C|D

  for (int s = 0; s < 4; ++s) {
    const float* base = ((s < 2) ? r0 : r1) + (s & 1) * 512;
    float x0 = base[t] * SCALE_;
    float x1 = base[t + 256] * SCALE_;
    red[t] = fmaxf(x0, x1);
    __syncthreads();
    for (int off = 128; off; off >>= 1) {
      if (t < off) red[t] = fmaxf(red[t], red[t + off]);
      __syncthreads();
    }
    float mx = red[0];
    __syncthreads();
    float e0 = __expf(x0 - mx), e1 = __expf(x1 - mx);
    red[t] = e0 + e1;
    __syncthreads();
    for (int off = 128; off; off >>= 1) {
      if (t < off) red[t] += red[t + off];
      __syncthreads();
    }
    float rs = 1.0f / red[0];
    __syncthreads();
    seg[s][t] = e0 * rs;
    seg[s][t + 256] = e1 * rs;
  }
  __syncthreads();

  long long obase = ((long long)b * 2048 + q) * 512;
  long long pbase = ((long long)b * 2048 + q) * 1024;
  for (int u = t; u < 512; u += 256) {
    float P = seg[0][u] - seg[3][u];   // A - D
    float S = seg[1][u] + seg[2][u];   // B + C
    attnR[obase + u] = P - S;
    attnI[obase + u] = P + S;
    PS[pbase + u] = (__bf16)P;
    PS[pbase + 512 + u] = (__bf16)S;
  }
}

// ---------------------------------------------------------------------------
extern "C" void kernel_launch(void* const* d_in, const int* in_sizes, int n_in,
                              void* d_out, int out_size, void* d_ws, size_t ws_size,
                              hipStream_t stream) {
  const float* Qr   = (const float*)d_in[0];
  const float* Qi   = (const float*)d_in[1];
  const float* W_Q  = (const float*)d_in[2];  const float* b_Q  = (const float*)d_in[3];
  const float* W_Qi = (const float*)d_in[4];  const float* b_Qi = (const float*)d_in[5];
  const float* W_K  = (const float*)d_in[6];  const float* b_K  = (const float*)d_in[7];
  const float* W_Ki = (const float*)d_in[8];  const float* b_Ki = (const float*)d_in[9];
  const float* W_V  = (const float*)d_in[10]; const float* b_V  = (const float*)d_in[11];
  const float* W_Vi = (const float*)d_in[12]; const float* b_Vi = (const float*)d_in[13];
  const float* W_o  = (const float*)d_in[14]; const float* b_o  = (const float*)d_in[15];
  float* out = (float*)d_out;
  float* finalR = out;
  float* finalI = out + 16777216LL;
  float* attnR  = out + 2 * 16777216LL;
  float* attnI  = out + 3 * 16777216LL;

  char* wsb = (char*)d_ws;
  size_t off = 0;
  auto carve = [&](size_t bytes) -> void* {
    void* p = wsb + off;
    off += (bytes + 255) & ~(size_t)255;
    return p;
  };
  __bf16* Xbf   = (__bf16*)carve(32768ULL * 1024 * 2);
  __bf16* Wproj = (__bf16*)carve(3072ULL * 1024 * 2);
  float*  biasP = (float*) carve(3072ULL * 4);
  __bf16* PROJ  = (__bf16*)carve(32768ULL * 3072 * 2);
  __bf16* KK    = (__bf16*)carve(16ULL * 1024 * 512 * 2);
  __bf16* VV    = (__bf16*)carve(16ULL * 1024 * 1024 * 2);
  float*  LOG   = (float*) carve(32ULL * 2048 * 1024 * 4);
  __bf16* PS    = (__bf16*)carve(16ULL * 2048 * 1024 * 2);
  __bf16* OB    = (__bf16*)carve(32768ULL * 1024 * 2);
  __bf16* Bt4   = (__bf16*)carve(1024ULL * 1024 * 2);
  float*  bias4 = (float*) carve(1024ULL * 4);

  // --- pack operands (independent) ---
  pack_x<<<131072, 256, 0, stream>>>(Qr, Qi, Xbf);
  pack_wproj<<<12288, 256, 0, stream>>>(W_Q, b_Q, W_Qi, b_Qi, W_K, b_K, W_Ki, b_Ki,
                                        W_V, b_V, W_Vi, b_Vi, Wproj, biasP);
  pack_wout<<<4096, 256, 0, stream>>>(W_o, b_o, Bt4, bias4);

  // --- G1: projections  [32768,1024] x [3072,1024]^T -> PROJ bf16 [32768,3072]
  GemmParams p1 = {};
  p1.A = Xbf; p1.B = Wproj; p1.outH = PROJ; p1.bias = biasP;
  p1.lda = 1024; p1.ldb = 1024; p1.ldc = 3072; p1.K = 1024;
  p1.zdiv = 1; p1.mode = 1;
  gemm_bf16_wmma<<<dim3(24, 256, 1), 256, 0, stream>>>(p1);

  // --- pool + complex mix -> KK, VV
  pool_combine<<<16384, 256, 0, stream>>>(PROJ, KK, VV);

  // --- G2: logits  per (b,ri): [2048,512] x [1024,512]^T -> LOG fp32
  GemmParams p2 = {};
  p2.A = PROJ; p2.B = KK; p2.outF = LOG; p2.bias = nullptr;
  p2.lda = 3072; p2.ldb = 512; p2.ldc = 1024; p2.K = 512;
  p2.zdiv = 2; p2.mode = 0;
  p2.aZ1 = 2048LL * 3072; p2.aZ2 = 512;
  p2.bZ1 = 1024LL * 512;  p2.bZ2 = 0;
  p2.cZ1 = 2LL * 2048 * 1024; p2.cZ2 = 2048LL * 1024;
  gemm_bf16_wmma<<<dim3(8, 16, 32), 256, 0, stream>>>(p2);

  // --- softmax + P/S combine; writes attn_r/attn_i to d_out, PS bf16
  softmax_combine<<<32768, 256, 0, stream>>>(LOG, attnR, attnI, PS);

  // --- G3: out GEMM  per b: [2048,1024] x [1024,1024]^T -> OB (shuffled bf16)
  GemmParams p3 = {};
  p3.A = PS; p3.B = VV; p3.outH = OB; p3.bias = nullptr;
  p3.lda = 1024; p3.ldb = 1024; p3.ldc = 1024; p3.K = 1024;
  p3.zdiv = 1; p3.mode = 2;
  p3.aZ1 = 2048LL * 1024; p3.bZ1 = 1024LL * 1024; p3.cZ1 = 2048LL * 1024;
  gemm_bf16_wmma<<<dim3(8, 16, 16), 256, 0, stream>>>(p3);

  // --- G4: final GEMM  [32768,1024] x [1024,1024]^T -> final_r/final_i fp32
  GemmParams p4 = {};
  p4.A = OB; p4.B = Bt4; p4.outF = finalR; p4.outF2 = finalI; p4.bias = bias4;
  p4.lda = 1024; p4.ldb = 1024; p4.ldc = 512; p4.K = 1024;
  p4.zdiv = 1; p4.mode = 3;
  gemm_bf16_wmma<<<dim3(8, 256, 1), 256, 0, stream>>>(p4);
}
